// QHNet_77068893160085
// MI455X (gfx1250) — matrix operand order repaired and checked by reading, not compiled, not run on hardware
//
#include <hip/hip_runtime.h>

// CDNA5 / gfx1250. wave32. BF16 WMMA 16x16x32 with hi/lo split-bf16 (3-term)
// for ~f32 accuracy while staying on the matrix pipe.
// Two-phase structure (x0-phase, x1-phase) keeps VGPR count < 256.

typedef __attribute__((ext_vector_type(16))) __bf16 v16bf;
typedef __attribute__((ext_vector_type(8)))  float  v8f;
typedef __attribute__((ext_vector_type(4)))  float  v4f;
typedef __attribute__((ext_vector_type(2)))  float  v2f;

#define NW_TOT 61440   // total weight elements (= packed bf16 per plane)

// ---------------------------------------------------------------------------
// Prep: scale weights, split into bf16 hi/lo, emit in WMMA B-fragment order.
// Fragment = 32x16 bf16 tile (K x N), 512 bf16 = 1024 B.
// lane l, elem j -> W[kt*32 + (l>>4)*8 + (j&7) + (j>>3)*16, nt*16 + (l&15)]
// frag index = base[g] + nt*nK[g] + kt
//   g0 W000: frags   0..63  (nN=16,nK=4) scale 1/128
//   g1 W011: frags  64..79  (nN=4, nK=4) scale 1/(128*sqrt3)
//   g2 W101: frags  80..95  (nN=8, nK=2) scale 1/(64*sqrt3)
//   g3 W110: frags  96..111 (nN=8, nK=2) scale 1/(64*sqrt3)
//   g4 W111: frags 112..119 (nN=4, nK=2) scale 1/(64*sqrt6)
// hi plane at pk[0..NW_TOT), lo plane at pk[NW_TOT..2*NW_TOT)
// ---------------------------------------------------------------------------
__global__ void qhnet_prep(const float* __restrict__ w,
                           unsigned short* __restrict__ pkraw) {
  __bf16* pk = (__bf16*)pkraw;
  int e = blockIdx.x * blockDim.x + threadIdx.x;
  if (e >= NW_TOT) return;
  int frag   = e >> 9;
  int within = e & 511;
  int lane   = within >> 4;
  int j      = within & 15;

  const float S3 = 1.7320508075688772f;
  const float S6 = 2.449489742783178f;
  int base, nK, ncols, off; float scale;
  if (frag < 64)       { base = 0;   nK = 4; ncols = 256; off = 0;     scale = 1.0f / 128.0f; }
  else if (frag < 80)  { base = 64;  nK = 4; ncols = 64;  off = 32768; scale = 1.0f / (128.0f * S3); }
  else if (frag < 96)  { base = 80;  nK = 2; ncols = 128; off = 40960; scale = 1.0f / (64.0f * S3); }
  else if (frag < 112) { base = 96;  nK = 2; ncols = 128; off = 49152; scale = 1.0f / (64.0f * S3); }
  else                 { base = 112; nK = 2; ncols = 64;  off = 57344; scale = 1.0f / (64.0f * S6); }

  int lf = frag - base;
  int nt = lf / nK;
  int kt = lf - nt * nK;
  int k  = kt * 32 + (lane >> 4) * 8 + (j & 7) + (j >> 3) * 16;
  int n  = nt * 16 + (lane & 15);

  float val = w[off + k * ncols + n] * scale;
  __bf16 hi = (__bf16)val;
  float rem = val - (float)hi;
  pk[e]          = hi;
  pk[NW_TOT + e] = (__bf16)rem;
}

// ---------------------------------------------------------------------------
// One 16x16 output tile: 3-term split-bf16 GEMM over NK K-steps of 32.
// ---------------------------------------------------------------------------
template <int NK>
static __device__ __forceinline__
v8f gemm_tile(const v16bf* __restrict__ pkH, const v16bf* __restrict__ pkL,
              int fragBase, int lane, const v16bf* ah, const v16bf* al) {
  v8f acc = {0.f, 0.f, 0.f, 0.f, 0.f, 0.f, 0.f, 0.f};
#pragma unroll
  for (int kt = 0; kt < NK; ++kt) {
    v16bf bh = pkH[(size_t)(fragBase + kt) * 32 + lane];
    v16bf bl = pkL[(size_t)(fragBase + kt) * 32 + lane];
    acc = __builtin_amdgcn_wmma_f32_16x16x32_bf16(false, ah[kt], false, bh, (short)0, acc, false, false);
    acc = __builtin_amdgcn_wmma_f32_16x16x32_bf16(false, ah[kt], false, bl, (short)0, acc, false, false);
    acc = __builtin_amdgcn_wmma_f32_16x16x32_bf16(false, al[kt], false, bh, (short)0, acc, false, false);
  }
  return acc;
}

// ---------------------------------------------------------------------------
// Three 16x16 tiles sharing the same B fragments (three A planes).
// Loads each B hi/lo fragment once, feeds 9 WMMAs per K-step.
// ---------------------------------------------------------------------------
template <int NK>
static __device__ __forceinline__
void gemm_tile3(const v16bf* __restrict__ pkH, const v16bf* __restrict__ pkL,
                int fragBase, int lane,
                const v16bf (*ah)[2], const v16bf (*al)[2], v8f acc[3]) {
#pragma unroll
  for (int k = 0; k < 3; ++k)
    acc[k] = (v8f){0.f, 0.f, 0.f, 0.f, 0.f, 0.f, 0.f, 0.f};
#pragma unroll
  for (int kt = 0; kt < NK; ++kt) {
    v16bf bh = pkH[(size_t)(fragBase + kt) * 32 + lane];
    v16bf bl = pkL[(size_t)(fragBase + kt) * 32 + lane];
#pragma unroll
    for (int k = 0; k < 3; ++k) {
      acc[k] = __builtin_amdgcn_wmma_f32_16x16x32_bf16(false, ah[k][kt], false, bh, (short)0, acc[k], false, false);
      acc[k] = __builtin_amdgcn_wmma_f32_16x16x32_bf16(false, ah[k][kt], false, bl, (short)0, acc[k], false, false);
      acc[k] = __builtin_amdgcn_wmma_f32_16x16x32_bf16(false, al[k][kt], false, bh, (short)0, acc[k], false, false);
    }
  }
}

// ---------------------------------------------------------------------------
// Main: one wave per 16-batch tile.
//   Phase A (x0 fragments): blk00 + blk11 diagonal (pure r011 term).
//   Phase B (x1 fragments): blk01, blk10, blk11 off-diagonals (pure eps term).
// The two term sets never touch the same output element, so the phases are
// fully independent and their A fragments never coexist in registers.
// ---------------------------------------------------------------------------
__global__ void __launch_bounds__(256)
qhnet_main(const float* __restrict__ x_in, const unsigned short* __restrict__ pkraw,
           float* __restrict__ out, int nTiles) {
  const v16bf* pkH = (const v16bf*)pkraw;
  const v16bf* pkL = pkH + (NW_TOT / 16);

  int lane = threadIdx.x & 31;
  int wave = threadIdx.x >> 5;
  int tile = blockIdx.x * (blockDim.x >> 5) + wave;
  if (tile >= nTiles) return;       // wave-uniform: EXEC stays all-ones

  int b0  = tile * 16;
  int h   = lane >> 4;
  int l15 = lane & 15;
  const float* xr = x_in + (size_t)(b0 + l15) * 320;

  // D layout: vgpr r, lane l -> batch row m = (l>>4)*8 + r, GEMM col = nt*16 + (l&15)
  float* obase = out + (size_t)(b0 + h * 8) * 1600;

  // ======================= Phase A : x0 GEMMs ===========================
  {
    // A fragments from x0 (16x128) : 4 K-steps, ISA A layout
    // (lanes 0-15 hold K {h8+0..7, 16+h8+0..7}, h = lane>>4)
    v16bf a0h[4], a0l[4];
#pragma unroll
    for (int kt = 0; kt < 4; ++kt) {
      int c0 = kt * 32 + h * 8;
      v4f q0 = *(const v4f*)(xr + c0);
      v4f q1 = *(const v4f*)(xr + c0 + 4);
      v4f q2 = *(const v4f*)(xr + c0 + 16);
      v4f q3 = *(const v4f*)(xr + c0 + 20);
      float f[16];
#pragma unroll
      for (int i = 0; i < 4; ++i) { f[i] = q0[i]; f[4+i] = q1[i]; f[8+i] = q2[i]; f[12+i] = q3[i]; }
#pragma unroll
      for (int i = 0; i < 16; ++i) {
        __bf16 hb = (__bf16)f[i];
        a0h[kt][i] = hb;
        a0l[kt][i] = (__bf16)(f[i] - (float)hb);
      }
    }

    // blk00 : out[u, v], u = nt, v = l15
    for (int nt = 0; nt < 16; ++nt) {
      v8f acc = gemm_tile<4>(pkH, pkL, nt * 4, lane, a0h, a0l);
#pragma unroll
      for (int r = 0; r < 8; ++r)
        obase[(size_t)r * 1600 + nt * 40 + l15] = acc[r];
    }

    // blk11 diagonal : out[16+u*3+i, 16+v*3+i] = r011[u,v]
    for (int nt = 0; nt < 4; ++nt) {
      v8f sA = gemm_tile<4>(pkH, pkL, 64 + nt * 4, lane, a0h, a0l);
      int ncol = nt * 16 + l15;
      int u = ncol >> 3, v = ncol & 7;
      int row0 = 16 + u * 3, col0 = 16 + v * 3;
#pragma unroll
      for (int r = 0; r < 8; ++r) {
        float* pr = obase + (size_t)r * 1600 + (size_t)row0 * 40 + col0;
        float s = sA[r];
        pr[0]  = s;
        pr[41] = s;
        pr[82] = s;
      }
    }
  }

  __builtin_amdgcn_sched_barrier(0);   // keep phase B loads out of phase A

  // ======================= Phase B : x1 GEMMs ===========================
  {
    // A fragments from x1 (k-interleaved stride 3) : 3 planes x 2 K-steps
    v16bf a1h[3][2], a1l[3][2];
#pragma unroll
    for (int kt = 0; kt < 2; ++kt) {
      float s0[24], s1[24];
      {
        int w0 = kt * 32 + h * 8;
        const v2f* p0 = (const v2f*)(xr + 128 + w0 * 3);
        const v2f* p1 = (const v2f*)(xr + 128 + (w0 + 16) * 3);
#pragma unroll
        for (int t = 0; t < 12; ++t) { v2f q = p0[t]; s0[2*t] = q[0]; s0[2*t+1] = q[1]; }
#pragma unroll
        for (int t = 0; t < 12; ++t) { v2f q = p1[t]; s1[2*t] = q[0]; s1[2*t+1] = q[1]; }
      }
#pragma unroll
      for (int k = 0; k < 3; ++k) {
#pragma unroll
        for (int i = 0; i < 8; ++i) {
          float f0 = s0[3*i + k], f1 = s1[3*i + k];
          __bf16 h0 = (__bf16)f0;
          a1h[k][kt][i]     = h0;
          a1l[k][kt][i]     = (__bf16)(f0 - (float)h0);
          __bf16 h1 = (__bf16)f1;
          a1h[k][kt][8 + i] = h1;
          a1l[k][kt][8 + i] = (__bf16)(f1 - (float)h1);
        }
      }
    }

    // blk01 : out[u, 16 + v*3 + j], (u,v) = (ncol>>3, ncol&7)
    for (int nt = 0; nt < 8; ++nt) {
      v8f acc[3];
      gemm_tile3<2>(pkH, pkL, 80 + nt * 2, lane, a1h, a1l, acc);
      int ncol = nt * 16 + l15;
      int u = ncol >> 3, v = ncol & 7;
#pragma unroll
      for (int r = 0; r < 8; ++r) {
        float* pr = obase + (size_t)r * 1600 + u * 40 + 16 + v * 3;
#pragma unroll
        for (int j = 0; j < 3; ++j) pr[j] = acc[j][r];
      }
    }

    // blk10 : out[16 + u*3 + i, v], u = nt, v = l15
    for (int nt = 0; nt < 8; ++nt) {
      v8f acc[3];
      gemm_tile3<2>(pkH, pkL, 96 + nt * 2, lane, a1h, a1l, acc);
#pragma unroll
      for (int i = 0; i < 3; ++i) {
#pragma unroll
        for (int r = 0; r < 8; ++r)
          obase[(size_t)r * 1600 + (16 + nt * 3 + i) * 40 + l15] = acc[i][r];
      }
    }

    // blk11 off-diagonals : Levi-Civita combination of c0,c1,c2
    for (int nt = 0; nt < 4; ++nt) {
      v8f c[3];
      gemm_tile3<2>(pkH, pkL, 112 + nt * 2, lane, a1h, a1l, c);
      int ncol = nt * 16 + l15;
      int u = ncol >> 3, v = ncol & 7;
      int row0 = 16 + u * 3, col0 = 16 + v * 3;
#pragma unroll
      for (int r = 0; r < 8; ++r) {
        float* pr = obase + (size_t)r * 1600 + (size_t)row0 * 40 + col0;
        float c0 = c[0][r], c1 = c[1][r], c2 = c[2][r];
        pr[1]  =  c2; pr[2]  = -c1;
        pr[40] = -c2; pr[42] =  c0;
        pr[80] =  c1; pr[81] = -c0;
      }
    }
  }
}

// ---------------------------------------------------------------------------
extern "C" void kernel_launch(void* const* d_in, const int* in_sizes, int n_in,
                              void* d_out, int out_size, void* d_ws, size_t ws_size,
                              hipStream_t stream) {
  const float* x_in    = (const float*)d_in[0];
  const float* weights = (const float*)d_in[1];
  float* out = (float*)d_out;
  unsigned short* pk = (unsigned short*)d_ws;   // needs 2*NW_TOT*2 = 245760 B

  int nBatch = in_sizes[0] / 320;               // B = 100000
  int nTiles = nBatch / 16;                     // 6250, exact

  qhnet_prep<<<(NW_TOT + 255) / 256, 256, 0, stream>>>(weights, pk);

  const int wavesPerBlock = 8;                  // 256 threads = 8 wave32
  int nBlocks = (nTiles + wavesPerBlock - 1) / wavesPerBlock;
  qhnet_main<<<nBlocks, 32 * wavesPerBlock, 0, stream>>>(x_in, pk, out, nTiles);
}
